// Seq2Seq_31190052504065
// MI455X (gfx1250) — matrix operand order repaired and checked by reading, not compiled, not run on hardware
//
#include <hip/hip_runtime.h>

typedef __attribute__((ext_vector_type(8)))  __bf16 v8bf;
typedef __attribute__((ext_vector_type(16))) __bf16 v16bf;
typedef __attribute__((ext_vector_type(8)))  float  v8f;

#define B_SZ 64
#define HID  256
#define G4   1024
#define TENC 256
#define TDEC 64
#define NN   16
#define FF   8
#define HPAD 264   // 256 + 8 bf16 pad to break LDS bank conflicts

// ---------------- WMMA fragment helpers (CDNA5 v_wmma_f32_16x16x32_bf16) ----
__device__ __forceinline__ v16bf bf_cat(v8bf lo, v8bf hi){
  return __builtin_shufflevector(lo, hi, 0,1,2,3,4,5,6,7,8,9,10,11,12,13,14,15);
}

// A fragment (M=16 x K=32 bf16): lane L -> row L%16 ; K half = L/16.
__device__ __forceinline__ v16bf load_a_frag(const __bf16* base, int rowStride,
                                             int rowBase, int kb, int L){
  const __bf16* p = base + (rowBase + (L & 15)) * rowStride + kb + ((L >> 4) << 3);
  v8bf lo = *(const v8bf*)p;
  v8bf hi = *(const v8bf*)(p + 16);
  return bf_cat(lo, hi);
}

// B fragment (K=32 x N=16 bf16) read from row-major W[N][K]:
__device__ __forceinline__ v16bf load_b_frag(const __bf16* W, int kStride,
                                             int nBase, int kb, int L){
  const __bf16* p = W + (size_t)(nBase + (L & 15)) * kStride + kb + ((L >> 4) << 3);
  v8bf lo = *(const v8bf*)p;
  v8bf hi = *(const v8bf*)(p + 16);
  return bf_cat(lo, hi);
}

__device__ __forceinline__ v8f wmma_bf16(v16bf a, v16bf b, v8f c){
  return __builtin_amdgcn_wmma_f32_16x16x32_bf16(false, a, false, b, (short)0, c, false, false);
}

__device__ __forceinline__ float sigm(float x){ return 1.0f / (1.0f + __expf(-x)); }

__device__ __forceinline__ v8f bcast8(float v){
  v8f r;
  #pragma unroll
  for(int i=0;i<8;i++) r[i]=v;
  return r;
}

// wave-private fragment-major c-state slot: 8 floats per (wave, m-tile, lane)
__device__ __forceinline__ int cfrag_off(int w, int m, int L){
  return (((w<<2) + m)*32 + L) << 3;
}

// ---------------- prep kernels ----------------------------------------------
__global__ void k_cvt_bf16(const float* __restrict__ s, __bf16* __restrict__ d, int n){
  int i = blockIdx.x*blockDim.x + threadIdx.x;
  if(i<n) d[i] = (__bf16)s[i];
}
__global__ void k_pad_wih(const float* __restrict__ s, __bf16* __restrict__ d){ // 1024x8 -> 1024x32
  int i = blockIdx.x*blockDim.x + threadIdx.x;
  if(i < G4*32){ int r=i>>5, c=i&31; d[i] = (c<FF) ? (__bf16)s[r*FF+c] : (__bf16)0.0f; }
}
__global__ void k_pad_fc(const float* __restrict__ s, __bf16* __restrict__ d){ // 8x256 -> 16x256
  int i = blockIdx.x*blockDim.x + threadIdx.x;
  if(i < 16*HID){ int r=i>>8, c=i&255; d[i] = (r<FF) ? (__bf16)s[(size_t)r*HID+c] : (__bf16)0.0f; }
}
__global__ void k_bias_sum(const float* __restrict__ a, const float* __restrict__ b,
                           float* __restrict__ d){
  int i = blockIdx.x*blockDim.x + threadIdx.x;
  if(i<G4) d[i]=a[i]+b[i];
}

// ---------------- encoder: one persistent WG per node -----------------------
__global__ void __launch_bounds__(512)
encoder_kernel(const float* __restrict__ x,
               const __bf16* __restrict__ Wih0p, const __bf16* __restrict__ Whh0,
               const __bf16* __restrict__ Wih1,  const __bf16* __restrict__ Whh1,
               const float* __restrict__ bsum0,  const float* __restrict__ bsum1,
               float* __restrict__ th, float* __restrict__ tc)
{
  const int tid  = threadIdx.x;
  const int w    = tid >> 5;
  const int L    = tid & 31;
  const int node = blockIdx.x;
  const int ln   = L & 15;
  const int hh   = L >> 4;

  extern __shared__ char smem[];
  __bf16* h0  = (__bf16*)smem;                 // [64][HPAD]
  __bf16* h1  = h0 + B_SZ * HPAD;              // [64][HPAD]
  __bf16* xb  = h1 + B_SZ * HPAD;              // [64][32]
  float*  c0f = (float*)(xb + B_SZ * 32);      // 16*4*32*8 f32 (frag-major c, layer0)
  float*  c1f = c0f + 16*4*32*8;               // layer1

  for(int i=tid;i<B_SZ*HPAD;i+=512){ h0[i]=(__bf16)0.0f; h1[i]=(__bf16)0.0f; }
  for(int i=tid;i<B_SZ*32;i+=512) xb[i]=(__bf16)0.0f;
  for(int i=tid;i<2*16*4*32*8;i+=512) c0f[i]=0.f;   // zero c0f and c1f

  float bias0[4], bias1[4];
  #pragma unroll
  for(int g=0;g<4;g++){
    bias0[g] = bsum0[g*256 + w*16 + ln];
    bias1[g] = bsum1[g*256 + w*16 + ln];
  }
  __syncthreads();

  #pragma unroll 1
  for(int t=0;t<TENC;t++){
    { // stage x_t (zero-padded to K=32)
      int b = tid >> 3, f = tid & 7;
      xb[b*32 + f] = (__bf16)x[ (((size_t)b*TENC + t)*NN + node)*FF + f ];
    }
    __syncthreads();

    // ---- layer 0: gates = h0@Whh0^T + x_t@Wih0^T + bias
    v8f acc[4][4];
    #pragma unroll
    for(int g=0;g<4;g++)
      #pragma unroll
      for(int m=0;m<4;m++) acc[g][m] = bcast8(bias0[g]);

    #pragma unroll 1
    for(int k=0;k<8;k++){
      v16bf a[4];
      #pragma unroll
      for(int m=0;m<4;m++) a[m] = load_a_frag(h0, HPAD, m*16, k*32, L);
      #pragma unroll
      for(int g=0;g<4;g++){
        v16bf bfrag = load_b_frag(Whh0, HID, (w+16*g)*16, k*32, L);
        #pragma unroll
        for(int m=0;m<4;m++) acc[g][m] = wmma_bf16(a[m], bfrag, acc[g][m]);
      }
    }
    { // input contribution, K=32 (cols 8..31 zero)
      v16bf a[4];
      #pragma unroll
      for(int m=0;m<4;m++) a[m] = load_a_frag(xb, 32, m*16, 0, L);
      #pragma unroll
      for(int g=0;g<4;g++){
        v16bf bfrag = load_b_frag(Wih0p, 32, (w+16*g)*16, 0, L);
        #pragma unroll
        for(int m=0;m<4;m++) acc[g][m] = wmma_bf16(a[m], bfrag, acc[g][m]);
      }
    }
    __syncthreads();   // all waves finished reading h0(t-1)

    #pragma unroll
    for(int m=0;m<4;m++){
      float* cp = c0f + cfrag_off(w, m, L);
      v8f c = *(v8f*)cp;
      #pragma unroll
      for(int v=0;v<8;v++){
        float gi=acc[0][m][v], gf=acc[1][m][v], gg=acc[2][m][v], go=acc[3][m][v];
        float cn = sigm(gf)*c[v] + sigm(gi)*tanhf(gg);
        c[v] = cn;
        float h = sigm(go)*tanhf(cn);
        int b = m*16 + (hh<<3) + v;
        h0[b*HPAD + w*16 + ln] = (__bf16)h;
      }
      *(v8f*)cp = c;
    }
    __syncthreads();

    // ---- layer 1: gates = h1@Whh1^T + h0(t)@Wih1^T + bias
    #pragma unroll
    for(int g=0;g<4;g++)
      #pragma unroll
      for(int m=0;m<4;m++) acc[g][m] = bcast8(bias1[g]);

    #pragma unroll 1
    for(int k=0;k<8;k++){
      v16bf a[4];
      #pragma unroll
      for(int m=0;m<4;m++) a[m] = load_a_frag(h1, HPAD, m*16, k*32, L);
      #pragma unroll
      for(int g=0;g<4;g++){
        v16bf bfrag = load_b_frag(Whh1, HID, (w+16*g)*16, k*32, L);
        #pragma unroll
        for(int m=0;m<4;m++) acc[g][m] = wmma_bf16(a[m], bfrag, acc[g][m]);
      }
    }
    #pragma unroll 1
    for(int k=0;k<8;k++){
      v16bf a[4];
      #pragma unroll
      for(int m=0;m<4;m++) a[m] = load_a_frag(h0, HPAD, m*16, k*32, L);
      #pragma unroll
      for(int g=0;g<4;g++){
        v16bf bfrag = load_b_frag(Wih1, HID, (w+16*g)*16, k*32, L);
        #pragma unroll
        for(int m=0;m<4;m++) acc[g][m] = wmma_bf16(a[m], bfrag, acc[g][m]);
      }
    }
    __syncthreads();

    #pragma unroll
    for(int m=0;m<4;m++){
      float* cp = c1f + cfrag_off(w, m, L);
      v8f c = *(v8f*)cp;
      #pragma unroll
      for(int v=0;v<8;v++){
        float gi=acc[0][m][v], gf=acc[1][m][v], gg=acc[2][m][v], go=acc[3][m][v];
        float cn = sigm(gf)*c[v] + sigm(gi)*tanhf(gg);
        c[v] = cn;
        float h = sigm(go)*tanhf(cn);
        int b = m*16 + (hh<<3) + v;
        h1[b*HPAD + w*16 + ln] = (__bf16)h;
      }
      *(v8f*)cp = c;
    }
    __syncthreads();
  }

  // write total_h [n][l][b][h] and total_c
  for(int i=tid;i<B_SZ*HID;i+=512){
    int b = i >> 8, j = i & 255;
    th[ (((size_t)node*2 + 0)*B_SZ + b)*HID + j ] = (float)h0[b*HPAD + j];
    th[ (((size_t)node*2 + 1)*B_SZ + b)*HID + j ] = (float)h1[b*HPAD + j];
  }
  #pragma unroll
  for(int m=0;m<4;m++){
    v8f c0 = *(v8f*)(c0f + cfrag_off(w, m, L));
    v8f c1 = *(v8f*)(c1f + cfrag_off(w, m, L));
    #pragma unroll
    for(int v=0;v<8;v++){
      int b = m*16 + (hh<<3) + v;
      int j = w*16 + ln;
      tc[ (((size_t)node*2 + 0)*B_SZ + b)*HID + j ] = c0[v];
      tc[ (((size_t)node*2 + 1)*B_SZ + b)*HID + j ] = c1[v];
    }
  }
}

// ---------------- attention (tiny, f32 VALU) --------------------------------
__global__ void attention_kernel(const float* __restrict__ th, const float* __restrict__ tc,
                                 const float* __restrict__ aW, const float* __restrict__ ab,
                                 float* __restrict__ ah, float* __restrict__ ac,
                                 const int* __restrict__ numt, const int* __restrict__ tord)
{
  int tid  = threadIdx.x;
  int j    = tid >> 8;
  int type = (tid >> 7) & 1;
  int l    = (tid >> 6) & 1;
  int b    = tid & 63;
  if (j >= *numt) return;
  int idx = *tord + j;
  const float* tot = type ? tc : th;
  float*       dst = type ? ac : ah;
  const float* wa = aW;
  const float* wb = aW + HID;
  const float* sel = tot + (((size_t)idx*2 + l)*B_SZ + b)*HID;
  float ssel = ab[0];
  for(int h=0;h<HID;h++) ssel += sel[h]*wa[h];
  float s[NN]; float mx = -1e30f;
  for(int n=0;n<NN;n++){
    const float* p = tot + (((size_t)n*2 + l)*B_SZ + b)*HID;
    float sc = ssel;
    for(int h=0;h<HID;h++) sc += p[h]*wb[h];
    s[n]=sc; mx = fmaxf(mx, sc);
  }
  float den=0.f;
  for(int n=0;n<NN;n++){ s[n]=__expf(s[n]-mx); den+=s[n]; }
  float inv = 1.f/den;
  float* d = dst + (((size_t)j*2 + l)*B_SZ + b)*HID;
  for(int h=0;h<HID;h++){
    float a = 0.f;
    for(int n=0;n<NN;n++) a += s[n]*tot[(((size_t)n*2+l)*B_SZ+b)*HID + h];
    d[h] = a*inv;
  }
}

// ---------------- decoder: one WG per target --------------------------------
__global__ void __launch_bounds__(512)
decoder_kernel(const float* __restrict__ x,
               const __bf16* __restrict__ Wih0p, const __bf16* __restrict__ Whh0,
               const __bf16* __restrict__ Wih1,  const __bf16* __restrict__ Whh1,
               const float* __restrict__ bsum0,  const float* __restrict__ bsum1,
               const __bf16* __restrict__ fcWp,  const float* __restrict__ fcb,
               const float* __restrict__ ah, const float* __restrict__ ac,
               float* __restrict__ Zws, float* __restrict__ out,
               const int* __restrict__ numt, const int* __restrict__ tord)
{
  const int j = blockIdx.x;
  if (j >= *numt) return;
  const int idx = *tord + j;
  const int tid = threadIdx.x;
  const int w   = tid >> 5;
  const int L   = tid & 31;
  const int ln  = L & 15;
  const int hh  = L >> 4;

  extern __shared__ char smem[];
  __bf16* hb0  = (__bf16*)smem;                // [64][HPAD]
  __bf16* hb1  = hb0 + B_SZ * HPAD;
  __bf16* ib   = hb1 + B_SZ * HPAD;            // [64][32]
  float*  nc0f = (float*)(ib + B_SZ * 32);     // frag-major attention c, layer0
  float*  nc1f = nc0f + 16*4*32*8;             // layer1

  // --- Z precompute: Z_l = new_h_l @ Whh_l^T + bias (constant over steps) ---
  for(int l=0;l<2;l++){
    const float* src = ah + (((size_t)j*2 + l)*B_SZ)*HID;
    for(int i=tid;i<B_SZ*HID;i+=512){ int b=i>>8,k=i&255; hb0[b*HPAD+k]=(__bf16)src[i]; }
    __syncthreads();
    const float* bs = l ? bsum1 : bsum0;
    v8f acc[4][4];
    #pragma unroll
    for(int g=0;g<4;g++){
      float bv = bs[g*256 + w*16 + ln];
      #pragma unroll
      for(int m=0;m<4;m++) acc[g][m] = bcast8(bv);
    }
    const __bf16* Wsrc = l ? Whh1 : Whh0;
    #pragma unroll 1
    for(int k=0;k<8;k++){
      v16bf a[4];
      #pragma unroll
      for(int m=0;m<4;m++) a[m] = load_a_frag(hb0, HPAD, m*16, k*32, L);
      #pragma unroll
      for(int g=0;g<4;g++){
        v16bf bfrag = load_b_frag(Wsrc, HID, (w+16*g)*16, k*32, L);
        #pragma unroll
        for(int m=0;m<4;m++) acc[g][m] = wmma_bf16(a[m], bfrag, acc[g][m]);
      }
    }
    float* Zl = Zws + (((size_t)j*2 + l)*B_SZ)*G4;
    #pragma unroll
    for(int g=0;g<4;g++)
      #pragma unroll
      for(int m=0;m<4;m++)
        *(v8f*)(Zl + (size_t)(((w*16 + g*4 + m)*32 + L))*8) = acc[g][m];
    __syncthreads();
  }

  // stage attention c into wave-private fragment-major LDS
  #pragma unroll
  for(int m=0;m<4;m++){
    v8f t0, t1;
    #pragma unroll
    for(int v=0;v<8;v++){
      int b = m*16 + (hh<<3) + v; int jj = w*16 + ln;
      t0[v] = ac[ (((size_t)j*2 + 0)*B_SZ + b)*HID + jj ];
      t1[v] = ac[ (((size_t)j*2 + 1)*B_SZ + b)*HID + jj ];
    }
    *(v8f*)(nc0f + cfrag_off(w, m, L)) = t0;
    *(v8f*)(nc1f + cfrag_off(w, m, L)) = t1;
  }
  float fcbias = (ln < 8) ? fcb[ln] : 0.f;

  // stage dec_in = x[:, -1, idx, :]
  for(int i=tid;i<B_SZ*32;i+=512) ib[i]=(__bf16)0.0f;
  {
    int b = tid >> 3, f = tid & 7;
    ib[b*32 + f] = (__bf16)x[ (((size_t)b*TENC + (TENC-1))*NN + idx)*FF + f ];
  }
  __syncthreads();

  const float* Z0 = Zws + (((size_t)j*2 + 0)*B_SZ)*G4;
  const float* Z1 = Zws + (((size_t)j*2 + 1)*B_SZ)*G4;

  #pragma unroll 1
  for(int t=0;t<TDEC;t++){
    v8f acc[4][4];
    // ---- cell 0: gates = inp@Wih0^T + Z0   (K = 32)
    #pragma unroll
    for(int g=0;g<4;g++)
      #pragma unroll
      for(int m=0;m<4;m++)
        acc[g][m] = *(const v8f*)(Z0 + (size_t)(((w*16 + g*4 + m)*32 + L))*8);
    {
      v16bf a[4];
      #pragma unroll
      for(int m=0;m<4;m++) a[m] = load_a_frag(ib, 32, m*16, 0, L);
      #pragma unroll
      for(int g=0;g<4;g++){
        v16bf bfrag = load_b_frag(Wih0p, 32, (w+16*g)*16, 0, L);
        #pragma unroll
        for(int m=0;m<4;m++) acc[g][m] = wmma_bf16(a[m], bfrag, acc[g][m]);
      }
    }
    #pragma unroll
    for(int m=0;m<4;m++){
      v8f nc = *(const v8f*)(nc0f + cfrag_off(w, m, L));
      #pragma unroll
      for(int v=0;v<8;v++){
        float gi=acc[0][m][v], gf=acc[1][m][v], gg=acc[2][m][v], go=acc[3][m][v];
        float c = sigm(gf)*nc[v] + sigm(gi)*tanhf(gg);
        float h = sigm(go)*tanhf(c);
        int b = m*16 + (hh<<3) + v;
        hb0[b*HPAD + w*16 + ln] = (__bf16)h;
      }
    }
    __syncthreads();

    // ---- cell 1: gates = h0n@Wih1^T + Z1   (K = 256)
    #pragma unroll
    for(int g=0;g<4;g++)
      #pragma unroll
      for(int m=0;m<4;m++)
        acc[g][m] = *(const v8f*)(Z1 + (size_t)(((w*16 + g*4 + m)*32 + L))*8);
    #pragma unroll 1
    for(int k=0;k<8;k++){
      v16bf a[4];
      #pragma unroll
      for(int m=0;m<4;m++) a[m] = load_a_frag(hb0, HPAD, m*16, k*32, L);
      #pragma unroll
      for(int g=0;g<4;g++){
        v16bf bfrag = load_b_frag(Wih1, HID, (w+16*g)*16, k*32, L);
        #pragma unroll
        for(int m=0;m<4;m++) acc[g][m] = wmma_bf16(a[m], bfrag, acc[g][m]);
      }
    }
    #pragma unroll
    for(int m=0;m<4;m++){
      v8f nc = *(const v8f*)(nc1f + cfrag_off(w, m, L));
      #pragma unroll
      for(int v=0;v<8;v++){
        float gi=acc[0][m][v], gf=acc[1][m][v], gg=acc[2][m][v], go=acc[3][m][v];
        float c = sigm(gf)*nc[v] + sigm(gi)*tanhf(gg);
        float h = sigm(go)*tanhf(c);
        int b = m*16 + (hh<<3) + v;
        hb1[b*HPAD + w*16 + ln] = (__bf16)h;
      }
    }
    __syncthreads();

    // ---- fc: out = h1n@fc_W^T + fc_b (waves 0..3, N padded 8->16)
    if (w < 4){
      v8f o = bcast8(fcbias);
      #pragma unroll 1
      for(int k=0;k<8;k++){
        v16bf a     = load_a_frag(hb1, HPAD, w*16, k*32, L);
        v16bf bfrag = load_b_frag(fcWp, HID, 0, k*32, L);
        o = wmma_bf16(a, bfrag, o);
      }
      if (ln < 8){
        #pragma unroll
        for(int v=0;v<8;v++){
          int b = w*16 + (hh<<3) + v;
          float val = o[v];
          out[ (((size_t)b*TDEC + t)*NN + j)*FF + ln ] = val;
          ib[b*32 + ln] = (__bf16)val;   // feed back as next step's input
        }
      }
    }
    __syncthreads();
  }
}

// ---------------- launch ----------------------------------------------------
extern "C" void kernel_launch(void* const* d_in, const int* in_sizes, int n_in,
                              void* d_out, int out_size, void* d_ws, size_t ws_size,
                              hipStream_t stream)
{
  (void)in_sizes; (void)n_in; (void)ws_size;
  const float* x     = (const float*)d_in[0];
  const float* eWih0 = (const float*)d_in[2];
  const float* eWhh0 = (const float*)d_in[3];
  const float* ebih0 = (const float*)d_in[4];
  const float* ebhh0 = (const float*)d_in[5];
  const float* eWih1 = (const float*)d_in[6];
  const float* eWhh1 = (const float*)d_in[7];
  const float* ebih1 = (const float*)d_in[8];
  const float* ebhh1 = (const float*)d_in[9];
  const float* dWih0 = (const float*)d_in[10];
  const float* dWhh0 = (const float*)d_in[11];
  const float* dbih0 = (const float*)d_in[12];
  const float* dbhh0 = (const float*)d_in[13];
  const float* dWih1 = (const float*)d_in[14];
  const float* dWhh1 = (const float*)d_in[15];
  const float* dbih1 = (const float*)d_in[16];
  const float* dbhh1 = (const float*)d_in[17];
  const float* fcW   = (const float*)d_in[18];
  const float* fcb   = (const float*)d_in[19];
  const float* aW    = (const float*)d_in[20];
  const float* ab    = (const float*)d_in[21];
  const int*   numt  = (const int*)d_in[22];
  const int*   tord  = (const int*)d_in[23];

  unsigned char* W = (unsigned char*)d_ws;
  size_t off = 0;
  auto alloc = [&](size_t bytes)->void* {
    void* p = W + off; off += (bytes + 255) & ~(size_t)255; return p;
  };
  __bf16* eWhh0b = (__bf16*)alloc((size_t)G4*HID*2);
  __bf16* eWih1b = (__bf16*)alloc((size_t)G4*HID*2);
  __bf16* eWhh1b = (__bf16*)alloc((size_t)G4*HID*2);
  __bf16* dWhh0b = (__bf16*)alloc((size_t)G4*HID*2);
  __bf16* dWih1b = (__bf16*)alloc((size_t)G4*HID*2);
  __bf16* dWhh1b = (__bf16*)alloc((size_t)G4*HID*2);
  __bf16* eWih0p = (__bf16*)alloc((size_t)G4*32*2);
  __bf16* dWih0p = (__bf16*)alloc((size_t)G4*32*2);
  __bf16* fcWp   = (__bf16*)alloc((size_t)16*HID*2);
  float*  bsum_e0 = (float*)alloc((size_t)G4*4);
  float*  bsum_e1 = (float*)alloc((size_t)G4*4);
  float*  bsum_d0 = (float*)alloc((size_t)G4*4);
  float*  bsum_d1 = (float*)alloc((size_t)G4*4);
  float*  th  = (float*)alloc((size_t)NN*2*B_SZ*HID*4);
  float*  tc  = (float*)alloc((size_t)NN*2*B_SZ*HID*4);
  float*  ah  = (float*)alloc((size_t)4*2*B_SZ*HID*4);
  float*  ac  = (float*)alloc((size_t)4*2*B_SZ*HID*4);
  float*  Zws = (float*)alloc((size_t)4*2*B_SZ*G4*4);

  const int thr = 256;
  const int nW  = G4*HID;
  k_cvt_bf16<<<(nW+thr-1)/thr, thr, 0, stream>>>(eWhh0, eWhh0b, nW);
  k_cvt_bf16<<<(nW+thr-1)/thr, thr, 0, stream>>>(eWih1, eWih1b, nW);
  k_cvt_bf16<<<(nW+thr-1)/thr, thr, 0, stream>>>(eWhh1, eWhh1b, nW);
  k_cvt_bf16<<<(nW+thr-1)/thr, thr, 0, stream>>>(dWhh0, dWhh0b, nW);
  k_cvt_bf16<<<(nW+thr-1)/thr, thr, 0, stream>>>(dWih1, dWih1b, nW);
  k_cvt_bf16<<<(nW+thr-1)/thr, thr, 0, stream>>>(dWhh1, dWhh1b, nW);
  k_pad_wih<<<(G4*32+thr-1)/thr, thr, 0, stream>>>(eWih0, eWih0p);
  k_pad_wih<<<(G4*32+thr-1)/thr, thr, 0, stream>>>(dWih0, dWih0p);
  k_pad_fc<<<(16*HID+thr-1)/thr, thr, 0, stream>>>(fcW, fcWp);
  k_bias_sum<<<(G4+thr-1)/thr, thr, 0, stream>>>(ebih0, ebhh0, bsum_e0);
  k_bias_sum<<<(G4+thr-1)/thr, thr, 0, stream>>>(ebih1, ebhh1, bsum_e1);
  k_bias_sum<<<(G4+thr-1)/thr, thr, 0, stream>>>(dbih0, dbhh0, bsum_d0);
  k_bias_sum<<<(G4+thr-1)/thr, thr, 0, stream>>>(dbih1, dbhh1, bsum_d1);

  // LDS: h0,h1 (2*33792) + xb/ib (4096) + two 64KB frag-major c regions
  const size_t lds = (size_t)(2*B_SZ*HPAD + B_SZ*32)*sizeof(__bf16)
                   + (size_t)2*16*4*32*8*sizeof(float);   // 202752 B
  encoder_kernel<<<NN, 512, lds, stream>>>(x, eWih0p, eWhh0b, eWih1b, eWhh1b,
                                           bsum_e0, bsum_e1, th, tc);
  attention_kernel<<<1, 1024, 0, stream>>>(th, tc, aW, ab, ah, ac, numt, tord);
  hipMemsetAsync(d_out, 0, (size_t)out_size*sizeof(float), stream);
  decoder_kernel<<<4, 512, lds, stream>>>(x, dWih0p, dWhh0b, dWih1b, dWhh1b,
                                          bsum_d0, bsum_d1, fcWp, fcb,
                                          ah, ac, Zws, (float*)d_out, numt, tord);
}
